// TestRNNCell_80221399154904
// MI455X (gfx1250) — matrix operand order repaired and checked by reading, not compiled, not run on hardware
//
#include <hip/hip_runtime.h>
#include <hip/hip_bf16.h>

// ---------------------------------------------------------------------------
// Elman RNN on gfx1250 (MI455X):
//   Phase 1: xk = x @ W  (32768x1024x1024 GEMM, bf16 WMMA, fp32 accum) -> d_out
//   Phase 2: ONE persistent kernel scans t=0..511:
//              out[:,t,:] = relu(out[:,t,:] + h @ U),  h' = bf16(out[:,t,:])
//            8 resident workgroups sync per step with an agent-scope atomic
//            counter + release/acquire fences (global_wb / global_inv).
//
// Weights converted once to bf16, swizzled FRAGMENT-CONTIGUOUS:
//   word(kpair, n): kpair = blk*16 + (j>>2)*8 + hi*4 + (j&3) (j=0..7), n = ng*16 + c
//   -> u32 index  (((blk*2 + hi)*64 + ng)*16 + c)*8 + j
// so each lane's 8-word WMMA B fragment (ISA 16-bit B layout: lanes 0-15 hold
// K=0..7,16..23, lanes 16-31 hold K=8..15,24..31, one K-pair per VGPR) is TWO b128 loads
// into consecutive VGPRs -- no cross-moves into the even-aligned WMMA blocks.
// ---------------------------------------------------------------------------

typedef __attribute__((ext_vector_type(16))) __bf16 v16bf;
typedef __attribute__((ext_vector_type(2)))  __bf16 v2bf;
typedef __attribute__((ext_vector_type(8)))  float  v8f;

union Frag16 { uint4 q[2]; unsigned w[8]; v16bf v; };

#if defined(__has_builtin)
#if __has_builtin(__builtin_amdgcn_cvt_pk_bf16_f32)
#define HAVE_CVT_PK_BF16 1
#endif
#if __has_builtin(__builtin_amdgcn_perm)
#define HAVE_PERM 1
#endif
#endif

// Fast fp32x2 -> packed bf16x2: round-half-up (add 0x8000) + high-half merge.
__device__ __forceinline__ unsigned pk_bf16(float lo, float hi) {
#ifdef HAVE_CVT_PK_BF16
  v2bf r = __builtin_amdgcn_cvt_pk_bf16_f32(lo, hi);
  return __builtin_bit_cast(unsigned, r);
#else
  unsigned ul = __builtin_bit_cast(unsigned, lo) + 0x8000u;
  unsigned uh = __builtin_bit_cast(unsigned, hi) + 0x8000u;
#ifdef HAVE_PERM
  return __builtin_amdgcn_perm(uh, ul, 0x07060302u);  // {uh[31:16], ul[31:16]}
#else
  return (uh & 0xFFFF0000u) | (ul >> 16);
#endif
#endif
}
__device__ __forceinline__ unsigned short bf16_1(float f) {
  return (unsigned short)((__builtin_bit_cast(unsigned, f) + 0x8000u) >> 16);
}

// fp32 [1024 x 1024] row-major -> bf16 fragment-contiguous swizzle (see top)
__global__ __launch_bounds__(256) void convert_w(const float* __restrict__ in,
                                                 unsigned* __restrict__ outw) {
  int tid = blockIdx.x * blockDim.x + threadIdx.x;   // 0 .. 512*1024-1
  int j   = tid & 7;
  int c   = (tid >> 3) & 15;
  int ng  = (tid >> 7) & 63;
  int hi  = (tid >> 13) & 1;
  int blk = tid >> 14;                               // 32-K block
  int kp  = blk * 16 + (j >> 2) * 8 + hi * 4 + (j & 3); // K-pair row
  int n   = ng * 16 + c;
  float a = in[(size_t)(2 * kp)     * 1024 + n];
  float b = in[(size_t)(2 * kp + 1) * 1024 + n];
  outw[tid] = pk_bf16(a, b);
}

__global__ __launch_bounds__(256) void zero_h(unsigned* __restrict__ h,
                                              unsigned* __restrict__ bar) {
  int i = blockIdx.x * blockDim.x + threadIdx.x;     // 64*512 words
  h[i] = 0u;
  if (i == 0) *bar = 0u;
}

// ---------------------------------------------------------------------------
// Phase 1: out[m,n] = sum_k x[m,k] * W[k,n]   (M=32768, N=1024, K=1024)
// Wave tile 32x64: 2 A frags x 4 B frags -> 8 WMMAs / k-step, 16 b128 loads.
// Block = 8 waves stacked along M (256x64). grid = (N/64, M/256) = (16, 128)
// ---------------------------------------------------------------------------
__global__ __launch_bounds__(256) void gemm_xk(const float* __restrict__ x,
                                               const unsigned* __restrict__ wbf,
                                               float* __restrict__ out) {
  const int lane    = threadIdx.x & 31;
  const int wave    = threadIdx.x >> 5;
  const int n0      = blockIdx.x * 64;
  const int rowBase = blockIdx.y * 256 + wave * 32;
  const int colLane = lane & 15;
  const int hi      = lane >> 4;
  const int kbA     = hi * 8;    // A: lanes 16-31 hold K=8..15 / 24..31

  const float* arow0 = x + (size_t)(rowBase + colLane) * 1024 + kbA;
  const float* arow1 = arow0 + (size_t)16 * 1024;
  // B fragment base for this lane: blk/ti offsets added in the loop.
  const unsigned* bbase =
      wbf + ((((size_t)hi * 64 + (n0 >> 4)) * 16) + colLane) * 8;

  v8f acc[8];
#pragma unroll
  for (int i = 0; i < 8; ++i) acc[i] = (v8f){};

  for (int k0 = 0; k0 < 1024; k0 += 32) {
    if (k0 + 128 < 1024) {
      __builtin_prefetch(arow0 + k0 + 128, 0, 1);    // global_prefetch_b8
      __builtin_prefetch(arow1 + k0 + 128, 0, 1);
    }

    // A fragments: two contiguous 8-float runs per lane -> 4x b128 each.
    Frag16 a0, a1;
#pragma unroll
    for (int h2 = 0; h2 < 2; ++h2) {                 // 0: K=kb+0..7, 1: +16
      const float4* p0 = (const float4*)(arow0 + k0 + h2 * 16);
      const float4* p1 = (const float4*)(arow1 + k0 + h2 * 16);
      float4 f0a = p0[0], f0b = p0[1];
      float4 f1a = p1[0], f1b = p1[1];
      a0.w[h2 * 4 + 0] = pk_bf16(f0a.x, f0a.y);
      a0.w[h2 * 4 + 1] = pk_bf16(f0a.z, f0a.w);
      a0.w[h2 * 4 + 2] = pk_bf16(f0b.x, f0b.y);
      a0.w[h2 * 4 + 3] = pk_bf16(f0b.z, f0b.w);
      a1.w[h2 * 4 + 0] = pk_bf16(f1a.x, f1a.y);
      a1.w[h2 * 4 + 1] = pk_bf16(f1a.z, f1a.w);
      a1.w[h2 * 4 + 2] = pk_bf16(f1b.x, f1b.y);
      a1.w[h2 * 4 + 3] = pk_bf16(f1b.z, f1b.w);
    }

    // B fragments: fragment-contiguous -> two b128 loads each, in place.
    Frag16 b[4];
    const unsigned* bk = bbase + (size_t)(k0 >> 5) * 16384;  // blk*2*64*16*8
#pragma unroll
    for (int ti = 0; ti < 4; ++ti) {
      const uint4* bp = (const uint4*)(bk + ti * 128);       // ng step = 16*8
      b[ti].q[0] = bp[0];
      b[ti].q[1] = bp[1];
    }

#pragma unroll
    for (int ti = 0; ti < 4; ++ti) {
      acc[ti]     = __builtin_amdgcn_wmma_f32_16x16x32_bf16(
          false, a0.v, false, b[ti].v, (short)0, acc[ti],     false, false);
      acc[4 + ti] = __builtin_amdgcn_wmma_f32_16x16x32_bf16(
          false, a1.v, false, b[ti].v, (short)0, acc[4 + ti], false, false);
    }
  }

#pragma unroll
  for (int mi = 0; mi < 2; ++mi) {
#pragma unroll
    for (int r = 0; r < 8; ++r) {
      // D layout: VGPR r, lanes 0-15 -> M=r, lanes 16-31 -> M=r+8; N = lane&15
      float* o = out + (size_t)(rowBase + mi * 16 + r + hi * 8) * 1024
                     + n0 + colLane;
      o[0]  = acc[mi * 4 + 0][r];
      o[16] = acc[mi * 4 + 1][r];
      o[32] = acc[mi * 4 + 2][r];
      o[48] = acc[mi * 4 + 3][r];
    }
  }
}

// ---------------------------------------------------------------------------
// Phase 2: persistent scan. 8 blocks x 8 waves = 64 wave-jobs
// (4 batch-tiles of 16 x 16 unit-groups of 64). One grid barrier per step.
// ---------------------------------------------------------------------------
__global__ __launch_bounds__(256) void rnn_scan(float* __restrict__ out,
                                                const unsigned* __restrict__ ubf,
                                                unsigned* __restrict__ hping,
                                                unsigned* __restrict__ hpong,
                                                unsigned* __restrict__ bar) {
  const int lane    = threadIdx.x & 31;
  const int wave    = threadIdx.x >> 5;
  const int job     = blockIdx.x * 8 + wave;   // 0..63
  const int rowBase = (job & 3) * 16;          // batch tile
  const int n0      = (job >> 2) * 64;         // unit group
  const int colLane = lane & 15;
  const int hi      = lane >> 4;
  const int kbA     = hi * 8;

  const unsigned* bbase =
      ubf + ((((size_t)hi * 64 + (n0 >> 4)) * 16) + colLane) * 8;

  for (int t = 0; t < 512; ++t) {
    const unsigned* hprev = (t & 1) ? hpong : hping;
    unsigned short* hnext = (unsigned short*)((t & 1) ? hping : hpong);
    // h is bf16 row-major [64][1024]: per-lane A words are contiguous u32.
    const unsigned* ha = hprev + (size_t)(rowBase + colLane) * 512 + (kbA >> 1);

    v8f acc0 = {}, acc1 = {}, acc2 = {}, acc3 = {};

    for (int k0 = 0; k0 < 1024; k0 += 32) {
      Frag16 a;
      {
        const uint4* haq = (const uint4*)(ha + (k0 >> 1));
        a.q[0] = haq[0];                       // K = kb+0,2,4,6 pairs
        a.q[1] = haq[2];                       // K = kb+16..22 pairs (+8 u32)
      }

      Frag16 b0, b1, b2, b3;
      const unsigned* bk = bbase + (size_t)(k0 >> 5) * 16384;
      {
        const uint4* bp0 = (const uint4*)(bk);
        const uint4* bp1 = (const uint4*)(bk + 128);
        const uint4* bp2 = (const uint4*)(bk + 256);
        const uint4* bp3 = (const uint4*)(bk + 384);
        b0.q[0] = bp0[0]; b0.q[1] = bp0[1];
        b1.q[0] = bp1[0]; b1.q[1] = bp1[1];
        b2.q[0] = bp2[0]; b2.q[1] = bp2[1];
        b3.q[0] = bp3[0]; b3.q[1] = bp3[1];
      }

      acc0 = __builtin_amdgcn_wmma_f32_16x16x32_bf16(false, a.v, false, b0.v,
                                                     (short)0, acc0, false, false);
      acc1 = __builtin_amdgcn_wmma_f32_16x16x32_bf16(false, a.v, false, b1.v,
                                                     (short)0, acc1, false, false);
      acc2 = __builtin_amdgcn_wmma_f32_16x16x32_bf16(false, a.v, false, b2.v,
                                                     (short)0, acc2, false, false);
      acc3 = __builtin_amdgcn_wmma_f32_16x16x32_bf16(false, a.v, false, b3.v,
                                                     (short)0, acc3, false, false);
    }

#pragma unroll
    for (int r = 0; r < 8; ++r) {
      int row = rowBase + r + hi * 8;              // batch index
      size_t obase = (size_t)row * (512u * 1024u) + (size_t)t * 1024u;
#pragma unroll
      for (int ti = 0; ti < 4; ++ti) {
        int col = n0 + ti * 16 + colLane;          // unit index
        float accv = (ti == 0) ? acc0[r] : (ti == 1) ? acc1[r]
                   : (ti == 2) ? acc2[r] : acc3[r];
        float v = out[obase + col] + accv;
        v = fmaxf(v, 0.0f);
        out[obase + col] = v;
        hnext[(size_t)row * 1024 + col] = bf16_1(v);
      }
    }

    // ---- device-wide barrier: release stores, count arrivals, acquire ----
    __builtin_amdgcn_fence(__ATOMIC_RELEASE, "agent");   // wb + wait stores
    __syncthreads();
    if (threadIdx.x == 0) {
      __hip_atomic_fetch_add(bar, 1u, __ATOMIC_RELAXED, __HIP_MEMORY_SCOPE_AGENT);
      const unsigned target = 8u * (unsigned)(t + 1);    // monotonic counter
      while (__hip_atomic_load(bar, __ATOMIC_RELAXED, __HIP_MEMORY_SCOPE_AGENT)
             < target)
        __builtin_amdgcn_s_sleep(2);
    }
    __syncthreads();
    __builtin_amdgcn_fence(__ATOMIC_ACQUIRE, "agent");   // inv for fresh h
  }
}

// ---------------------------------------------------------------------------
extern "C" void kernel_launch(void* const* d_in, const int* in_sizes, int n_in,
                              void* d_out, int out_size, void* d_ws, size_t ws_size,
                              hipStream_t stream) {
  (void)in_sizes; (void)n_in; (void)out_size; (void)ws_size;

  const float* x   = (const float*)d_in[0];   // [64,512,1024] fp32
  const float* w   = (const float*)d_in[1];   // [1024,1024]   fp32
  const float* rec = (const float*)d_in[2];   // [1024,1024]   fp32
  float* out = (float*)d_out;                 // [64,512,1024] fp32

  unsigned char* ws = (unsigned char*)d_ws;
  unsigned* wbf = (unsigned*)ws;                                   // 2 MB
  unsigned* ubf = (unsigned*)(ws + (size_t)512 * 1024 * 4);        // 2 MB
  unsigned* h0  = (unsigned*)(ws + (size_t)2 * 512 * 1024 * 4);    // 128 KB
  unsigned* h1  = h0 + 64 * 512;                                   // 128 KB
  unsigned* bar = h1 + 64 * 512;                                   // 4 B

  convert_w<<<2048, 256, 0, stream>>>(w, wbf);
  convert_w<<<2048, 256, 0, stream>>>(rec, ubf);
  zero_h<<<128, 256, 0, stream>>>(h0, bar);

  gemm_xk<<<dim3(16, 128), 256, 0, stream>>>(x, wbf, out);

  rnn_scan<<<8, 256, 0, stream>>>(out, ubf, h0, h1, bar);
}
